// DotProductAttention_44976897524594
// MI455X (gfx1250) — matrix operand order repaired and checked by reading, not compiled, not run on hardware
//
#include <hip/hip_runtime.h>

typedef __attribute__((ext_vector_type(16))) _Float16 v16h;
typedef __attribute__((ext_vector_type(8)))  _Float16 v8h;
typedef __attribute__((ext_vector_type(8)))  float    v8f;

namespace {
constexpr int B_ = 16;
constexpr int L_ = 2048;
constexpr int D_ = 64;
constexpr int QW = 32;    // queries per wave (two 16-row tiles, shared A-fragments)
constexpr int KT = 32;    // keys per block iteration
constexpr int WAVES = 4;  // waves per block (share one K/V tile via LDS)
constexpr int KSTR = 72;  // LDS row stride in halves for K tile (16B-aligned, bank-friendly)
constexpr int VSTR = 40;  // LDS row stride in halves for V^T tile
// softmax in exp2 domain: fold 1/sqrt(D) * log2(e) into Q
constexpr float SCALE2 = 0.125f * 1.4426950408889634f;
}

// raw v_exp_f32 (2^x, flush-below-range -> 0: exactly right for softmax tails)
__device__ __forceinline__ float fast_exp2(float x) {
#if __has_builtin(__builtin_amdgcn_exp2f)
  return __builtin_amdgcn_exp2f(x);
#else
  return __builtin_exp2f(x);
#endif
}

// B-operand fragment from global: slot e holds p[e] (16 contiguous f32 -> f16).
__device__ __forceinline__ v16h frag_b16(const float* p, float s) {
  const float4* q = reinterpret_cast<const float4*>(p);
  float4 f0 = q[0], f1 = q[1], f2 = q[2], f3 = q[3];
  float t[16] = {f0.x,f0.y,f0.z,f0.w, f1.x,f1.y,f1.z,f1.w,
                 f2.x,f2.y,f2.z,f2.w, f3.x,f3.y,f3.z,f3.w};
  v16h r;
#pragma unroll
  for (int e = 0; e < 16; ++e) r[e] = (_Float16)(t[e] * s);
  return r;
}

// A-operand fragment from LDS (f16, 16x32 A layout):
// slot e -> p[base+e] (contiguous), base = 16*(lane>>4).
__device__ __forceinline__ v16h frag_lds(const _Float16* p, int base) {
  v8h lo = *reinterpret_cast<const v8h*>(p + base);        // ds_load_b128
  v8h hi = *reinterpret_cast<const v8h*>(p + base + 8);    // ds_load_b128
  v16h r;
#pragma unroll
  for (int e = 0; e < 8; ++e) { r[e] = lo[e]; r[e + 8] = hi[e]; }
  return r;
}

struct Tile {
  float4 k[4];
  float4 v[4];
};

__device__ __forceinline__ void load_tile(const float* kp, const float* vp, Tile& t) {
  const float4* ks = reinterpret_cast<const float4*>(kp);
  const float4* vs = reinterpret_cast<const float4*>(vp);
#pragma unroll
  for (int i = 0; i < 4; ++i) { t.k[i] = ks[i]; t.v[i] = vs[i]; }
}

__device__ __forceinline__ void store_tile(_Float16* KfB, _Float16* VtB,
                                           const Tile& t, int srow, int sseg) {
  float kt[16] = {t.k[0].x,t.k[0].y,t.k[0].z,t.k[0].w, t.k[1].x,t.k[1].y,t.k[1].z,t.k[1].w,
                  t.k[2].x,t.k[2].y,t.k[2].z,t.k[2].w, t.k[3].x,t.k[3].y,t.k[3].z,t.k[3].w};
  float wt[16] = {t.v[0].x,t.v[0].y,t.v[0].z,t.v[0].w, t.v[1].x,t.v[1].y,t.v[1].z,t.v[1].w,
                  t.v[2].x,t.v[2].y,t.v[2].z,t.v[2].w, t.v[3].x,t.v[3].y,t.v[3].z,t.v[3].w};
  v8h klo, khi;
#pragma unroll
  for (int e = 0; e < 8; ++e) {
    klo[e] = (_Float16)kt[e];
    khi[e] = (_Float16)kt[e + 8];
  }
  *reinterpret_cast<v8h*>(&KfB[srow * KSTR + sseg])     = klo; // ds_store_b128
  *reinterpret_cast<v8h*>(&KfB[srow * KSTR + sseg + 8]) = khi; // ds_store_b128
#pragma unroll
  for (int e = 0; e < 16; ++e)
    VtB[(size_t)(sseg + e) * VSTR + srow] = (_Float16)wt[e];    // transpose into LDS
}

__global__ __launch_bounds__(WAVES * 32)
void attn_fwd_wmma(const float* __restrict__ Q, const float* __restrict__ K,
                   const float* __restrict__ V, float* __restrict__ O) {
  // double-buffered staged tiles (f16): ~19 KB of the 320 KB/WGP
  __shared__ __align__(16) _Float16 Kf[2][32 * KSTR]; // K tile, row-major (key, d)
  __shared__ __align__(16) _Float16 Vt[2][64 * VSTR]; // V tile, transposed (d, key)

  const int tid  = threadIdx.x;
  const int lane = tid & 31;
  const int wave = tid >> 5;
  const int pid  = blockIdx.x * WAVES + wave;      // query-tile-pair id (exact grid)
  const int b    = pid / (L_ / QW);
  const int q0   = (pid % (L_ / QW)) * QW;
  const int n    = lane & 15;   // this lane's query within each 16-row tile
  const int h    = lane >> 4;   // half-wave select
  const int base = 16 * h;      // A-layout K-chunk base

  // Permuted key-row map for the S^T A-fragments so that the C-layout result
  // lands exactly in P^T's B-operand layout (no cross-lane exchange):
  //   block0 rows hold keys {0..7, 16..23}, block1 rows keys {8..15, 24..31}
  const int key0 = n + 8 * (n >> 3);  // block 0 key row for this lane
  const int key1 = key0 + 8;          // block 1 key row for this lane

  // cooperative-staging coordinates: 4 threads per row, 16 floats each
  const int srow = tid >> 2;          // 0..31 (key row)
  const int sseg = (tid & 3) * 16;    // d segment start

  // Q^T as B-operand, two query groups g: rows q0 + 16g + n.
  v16h bq[2][2];
#pragma unroll
  for (int g = 0; g < 2; ++g) {
    const float* qrow = Q + ((size_t)b * L_ + q0 + 16 * g + n) * D_;
    bq[g][0] = frag_b16(qrow + 16 * h,      SCALE2);
    bq[g][1] = frag_b16(qrow + 32 + 16 * h, SCALE2);
  }

  float m_run[2] = {-1e30f, -1e30f};  // running row max (exp2 domain) per group
  float l_run[2] = {0.0f, 0.0f};      // running row sum per group
  v8f acc[2][4] = {};                 // O^T accumulators per group, d chunk c

  const float* kbase = K + (size_t)b * L_ * D_ + (size_t)srow * D_ + sseg;
  const float* vbase = V + (size_t)b * L_ * D_ + (size_t)srow * D_ + sseg;

  // ---- software pipeline prologue: stage tile 0 into buffer 0 ----
  Tile treg;
  load_tile(kbase, vbase, treg);
  store_tile(Kf[0], Vt[0], treg, srow, sseg);

  for (int kb = 0; kb < L_; kb += KT) {
    const int cur = (kb >> 5) & 1;

    // issue next tile's global loads early (latency hidden under barrier+compute)
    const bool more = (kb + KT) < L_;
    if (more) load_tile(kbase + (size_t)(kb + KT) * D_, vbase + (size_t)(kb + KT) * D_, treg);
    if (kb + 2 * KT < L_) {  // warm L2/L0 for the tile after that
      __builtin_prefetch(kbase + (size_t)(kb + 2 * KT) * D_, 0, 3);
      __builtin_prefetch(vbase + (size_t)(kb + 2 * KT) * D_, 0, 3);
    }

    __syncthreads(); // buffer `cur` stores (prev iter) visible; prev reads of cur^1 done

    // ---- shared A-fragments: K rows (permuted), reused by both query groups ----
    const _Float16* KfC = Kf[cur];
    v16h ak00 = frag_lds(&KfC[key0 * KSTR + 0],  base);
    v16h ak01 = frag_lds(&KfC[key0 * KSTR + 32], base);
    v16h ak10 = frag_lds(&KfC[key1 * KSTR + 0],  base);
    v16h ak11 = frag_lds(&KfC[key1 * KSTR + 32], base);

    v16h bp[2];
#pragma unroll
    for (int g = 0; g < 2; ++g) {
      // ---- S^T = K_perm x Q_g^T (K-depth 64 = 2x32 chained) ----
      v8f c0 = {};
      c0 = __builtin_amdgcn_wmma_f32_16x16x32_f16(false, ak00, false, bq[g][0], (short)0, c0, false, false);
      c0 = __builtin_amdgcn_wmma_f32_16x16x32_f16(false, ak01, false, bq[g][1], (short)0, c0, false, false);
      v8f c1 = {};
      c1 = __builtin_amdgcn_wmma_f32_16x16x32_f16(false, ak10, false, bq[g][0], (short)0, c1, false, false);
      c1 = __builtin_amdgcn_wmma_f32_16x16x32_f16(false, ak11, false, bq[g][1], (short)0, c1, false, false);
      // lane (n,h): c0[r] = score(key 16h+r), c1[r] = score(key 16h+8+r)

      // ---- online softmax (exp2 domain); lane owns query q0+16g+n ----
      float mt = -1e30f;
#pragma unroll
      for (int r = 0; r < 8; ++r) mt = fmaxf(mt, fmaxf(c0[r], c1[r]));
      mt = fmaxf(mt, __shfl_xor(mt, 16, 32));

      // rescale only when the running max actually grew anywhere in the wave
      // (if no lane grew, corr == 2^0 == 1 exactly -> skipping is bit-exact)
      if (__ballot(mt > m_run[g]) != 0ull) {
        const float m_new = fmaxf(m_run[g], mt);
        const float corr  = fast_exp2(m_run[g] - m_new);
        l_run[g] *= corr;
#pragma unroll
        for (int c = 0; c < 4; ++c)
#pragma unroll
          for (int r = 0; r < 8; ++r) acc[g][c][r] *= corr;
        m_run[g] = m_new;
      }

      // P^T B-operand builds directly: slot e = key 16h+e, slot e+8 = key 16h+8+e
      float rs = 0.0f;
#pragma unroll
      for (int r = 0; r < 8; ++r) {
        float p0 = fast_exp2(c0[r] - m_run[g]);
        float p1 = fast_exp2(c1[r] - m_run[g]);
        rs += p0 + p1;
        bp[g][r]     = (_Float16)p0;
        bp[g][r + 8] = (_Float16)p1;
      }
      rs += __shfl_xor(rs, 16, 32);
      l_run[g] += rs;
    }

    // ---- O^T += V^T x P_g^T : each V^T chunk loaded once, used by both groups ----
    const _Float16* VtC = Vt[cur];
#pragma unroll
    for (int c = 0; c < 4; ++c) {
      v16h av = frag_lds(&VtC[(size_t)(16 * c + n) * VSTR], base);
      acc[0][c] = __builtin_amdgcn_wmma_f32_16x16x32_f16(false, av, false, bp[0],
                                                         (short)0, acc[0][c], false, false);
      acc[1][c] = __builtin_amdgcn_wmma_f32_16x16x32_f16(false, av, false, bp[1],
                                                         (short)0, acc[1][c], false, false);
    }

    // ---- stage next tile into the other buffer (made visible by next sync) ----
    if (more) store_tile(Kf[cur ^ 1], Vt[cur ^ 1], treg, srow, sseg);
  }

  // ---- epilogue: O[q][d], d = 16c + r + 8h -> 8 contiguous floats per chunk ----
#pragma unroll
  for (int g = 0; g < 2; ++g) {
    const float inv = 1.0f / l_run[g];
    float* orow = O + ((size_t)b * L_ + q0 + 16 * g + n) * D_;
#pragma unroll
    for (int c = 0; c < 4; ++c) {
      float4 lo = make_float4(acc[g][c][0]*inv, acc[g][c][1]*inv,
                              acc[g][c][2]*inv, acc[g][c][3]*inv);
      float4 hi = make_float4(acc[g][c][4]*inv, acc[g][c][5]*inv,
                              acc[g][c][6]*inv, acc[g][c][7]*inv);
      float4* dst = reinterpret_cast<float4*>(orow + 16 * c + 8 * h);
      dst[0] = lo;
      dst[1] = hi;
    }
  }
}

extern "C" void kernel_launch(void* const* d_in, const int* in_sizes, int n_in,
                              void* d_out, int out_size, void* d_ws, size_t ws_size,
                              hipStream_t stream) {
  const float* q = (const float*)d_in[0];
  const float* k = (const float*)d_in[1];
  const float* v = (const float*)d_in[2];
  float* out = (float*)d_out;
  (void)in_sizes; (void)n_in; (void)out_size; (void)d_ws; (void)ws_size;

  dim3 block(WAVES * 32);                  // 4 waves = 128 threads
  dim3 grid((B_ * (L_ / QW)) / WAVES);     // 1024 query-tile pairs / 4 waves = 256 blocks
  attn_fwd_wmma<<<grid, block, 0, stream>>>(q, k, v, out);
}
